// AE_14783277432849
// MI455X (gfx1250) — matrix operand order repaired and checked by reading, not compiled, not run on hardware
//
#include <hip/hip_runtime.h>
#include <hip/hip_bf16.h>
#include <math.h>

typedef __attribute__((ext_vector_type(16))) _Float16 v16h;
typedef __attribute__((ext_vector_type(8)))  float    v8f;

#define BB 16   // batch size (fixed by reference)
#define KCH 6   // Chebyshev order (fixed by reference)

static inline long cdivl(long a, long b) { return (a + b - 1) / b; }

// ---------------------------------------------------------------------------
// Setup kernels: degree/normalization + CSR-by-destination build
// ---------------------------------------------------------------------------
__global__ void k_zero_f(float* __restrict__ p, long n) {
    long i = (long)blockIdx.x * blockDim.x + threadIdx.x;
    if (i < n) p[i] = 0.f;
}

__global__ void k_zero_i(int* __restrict__ p, long n) {
    long i = (long)blockIdx.x * blockDim.x + threadIdx.x;
    if (i < n) p[i] = 0;
}

__global__ void k_deg(float* __restrict__ deg, const int* __restrict__ src, int E) {
    int e = blockIdx.x * blockDim.x + threadIdx.x;
    if (e < E) atomicAdd(&deg[src[e]], 1.f);
}

__global__ void k_edge_norm(float* __restrict__ nrm, const float* __restrict__ deg,
                            const int* __restrict__ src, const int* __restrict__ dst, int E) {
    int e = blockIdx.x * blockDim.x + threadIdx.x;
    if (e < E) {
        float ds = deg[src[e]], dd = deg[dst[e]];
        float gs = ds > 0.f ? rsqrtf(fmaxf(ds, 1.f)) : 0.f;
        float gd = dd > 0.f ? rsqrtf(fmaxf(dd, 1.f)) : 0.f;
        nrm[e] = -gs * gd;
    }
}

__global__ void k_cnt(int* __restrict__ cnt, const int* __restrict__ dst, int E) {
    int e = blockIdx.x * blockDim.x + threadIdx.x;
    if (e < E) atomicAdd(&cnt[dst[e]], 1);
}

// Exclusive scan, single thread (N <= 16384: negligible)
__global__ void k_scan(const int* __restrict__ cnt, int* __restrict__ rowptr, int N) {
    if (threadIdx.x == 0 && blockIdx.x == 0) {
        int acc = 0;
        for (int n = 0; n < N; ++n) { rowptr[n] = acc; acc += cnt[n]; }
        rowptr[N] = acc;
    }
}

__global__ void k_copy_i(int* __restrict__ dst, const int* __restrict__ src, long n) {
    long i = (long)blockIdx.x * blockDim.x + threadIdx.x;
    if (i < n) dst[i] = src[i];
}

// Place src index + edge norm into destination-sorted order
__global__ void k_place(int* __restrict__ cursor, const int* __restrict__ src,
                        const int* __restrict__ dst, const float* __restrict__ nrm,
                        int* __restrict__ srcP, float* __restrict__ nrmP, int E) {
    int e = blockIdx.x * blockDim.x + threadIdx.x;
    if (e < E) {
        int slot = atomicAdd(&cursor[dst[e]], 1);
        srcP[slot] = src[e];
        nrmP[slot] = nrm[e];
    }
}

// ---------------------------------------------------------------------------
// Chebyshev basis buffer laid out [rows, S], S = 6*Fin; block k holds Tx_k.
// ---------------------------------------------------------------------------
__global__ void k_block_copy(float* __restrict__ tx, const float* __restrict__ src,
                             long rows, int F, int S, int co) {
    long i = (long)blockIdx.x * blockDim.x + threadIdx.x;
    if (i >= rows * (long)F) return;
    long row = i / F; int f = (int)(i % F);
    tx[row * S + co + f] = src[i];
}

// Gather SpMM + fused recurrence (NO atomics):
//   tx[b,n,cout+f] = scale * sum_j nrmP[j]*tx[b,srcP[j],cin+f]  - tx[b,n,csub+f]
// (csub < 0 disables the subtraction; j ranges over CSR row n)
__global__ void k_spmm_gather(float* __restrict__ tx, const int* __restrict__ rowptr,
                              const int* __restrict__ srcP, const float* __restrict__ nrmP,
                              int N, int F, int S, int cin, int cout, int csub, float scale) {
    long tid = (long)blockIdx.x * blockDim.x + threadIdx.x;
    long total = (long)BB * N * F;
    if (tid >= total) return;
    int f = (int)(tid % F);
    long t = tid / F;
    int n = (int)(t % N);
    int b = (int)(t / N);
    const int beg = rowptr[n], end = rowptr[n + 1];
    float acc = 0.f;
    for (int j = beg; j < end; ++j) {
        acc += nrmP[j] * tx[((size_t)b * N + srcP[j]) * S + cin + f];
    }
    float v = scale * acc;
    if (csub >= 0) v -= tx[((size_t)b * N + n) * S + csub + f];
    tx[((size_t)b * N + n) * S + cout + f] = v;
}

// ---------------------------------------------------------------------------
// Structured pooling (3 nnz/row), gather-only; output row stride So can drop
// the result directly into block 0 of the next conv's basis buffer.
// ---------------------------------------------------------------------------
__global__ void k_pool(float* __restrict__ out, const float* __restrict__ x,
                       const int* __restrict__ cols, const float* __restrict__ vals,
                       int Nout, int Nin, int F, int So) {
    long tid = (long)blockIdx.x * blockDim.x + threadIdx.x;
    long total = (long)BB * Nout * F;
    if (tid >= total) return;
    int f = (int)(tid % F);
    long t = tid / F;
    int r = (int)(t % Nout);
    int b = (int)(t / Nout);
    float acc = 0.f;
#pragma unroll
    for (int j = 0; j < 3; ++j) {
        int c = cols[3 * r + j];
        acc += vals[3 * r + j] * x[((size_t)b * Nin + c) * F + f];
    }
    out[((size_t)b * Nout + r) * So + f] = acc;
}

// ---------------------------------------------------------------------------
// Weight repack: f32 [Fin, Fout] -> f16 fragments in exact WMMA B layout.
// Zero-padded at K/N edges so GEMM B loads need no guards at all.
// ---------------------------------------------------------------------------
__global__ void k_pack_w(_Float16* __restrict__ dst, const float* __restrict__ W,
                         int Fin, int Fout, int nct, long total) {
    long t = (long)blockIdx.x * blockDim.x + threadIdx.x;
    if (t >= total) return;
    int i    = (int)(t & 15);
    int lane = (int)((t >> 4) & 31);
    long rest = t >> 9;
    int ct = (int)(rest % nct);
    int ks = (int)(rest / nct);
    int kb  = ks * 32 + (i & 7) + 8 * (lane >> 4) + 16 * (i >> 3);
    int col = ct * 16 + (lane & 15);
    float v = (kb < Fin && col < Fout) ? W[(size_t)kb * Fout + col] : 0.f;
    dst[t] = (_Float16)v;
}

// ---------------------------------------------------------------------------
// WMMA GEMM (single pass, write-only epilogue):
//   out[rows,Fout] = A[rows,Fin] @ W[Fin,Fout] + bias (+ optional ELU)
// One wave per (16-row tile x NT 16-col tiles); A fragment reused across NT.
// A: 4x b128 loads per K-step; B: one 32B contiguous v16h load per fragment.
// Fragment layouts per CDNA5 ISA 7.12.2 (wave32).
// ---------------------------------------------------------------------------
template <int NT, bool GK, bool GN>
__global__ void k_wmma_gemm(const float* __restrict__ A, const _Float16* __restrict__ Bp,
                            const float* __restrict__ bias, float* __restrict__ out,
                            int rows, int Fin, int Fout, int nct, int elu) {
    const int lane = threadIdx.x;      // 0..31
    const int hi   = lane >> 4;        // lane half
    const int ln   = lane & 15;
    const int rowBase  = blockIdx.x * 16;
    const int ctBase   = blockIdx.y * NT;

    v8f c[NT];
#pragma unroll
    for (int nt = 0; nt < NT; ++nt) c[nt] = v8f{};

    const float* aRow = A + (size_t)(rowBase + ln) * Fin;
    const int ksteps = (Fin + 31) / 32;

    for (int ks = 0; ks < ksteps; ++ks) {
        // A fragment: lanes 0-15 hold K={0..7,16..23}, lanes 16-31 K+8.
        v16h a;
        if (!GK) {
#pragma unroll
            for (int rrun = 0; rrun < 2; ++rrun) {
                const int base = ks * 32 + rrun * 16 + 8 * hi;
                const float4 q0 = *reinterpret_cast<const float4*>(aRow + base);
                const float4 q1 = *reinterpret_cast<const float4*>(aRow + base + 4);
                a[8 * rrun + 0] = (_Float16)q0.x; a[8 * rrun + 1] = (_Float16)q0.y;
                a[8 * rrun + 2] = (_Float16)q0.z; a[8 * rrun + 3] = (_Float16)q0.w;
                a[8 * rrun + 4] = (_Float16)q1.x; a[8 * rrun + 5] = (_Float16)q1.y;
                a[8 * rrun + 6] = (_Float16)q1.z; a[8 * rrun + 7] = (_Float16)q1.w;
            }
        } else {
#pragma unroll
            for (int j = 0; j < 8; ++j) {
                int ka = ks * 32 + ((j < 4) ? 2 * j : 16 + 2 * (j - 4)) + 8 * hi;
                float x0 = aRow[(ka     < Fin) ? ka     : 0];
                float x1 = aRow[(ka + 1 < Fin) ? ka + 1 : 0];
                a[2 * j]     = (_Float16)((ka     < Fin) ? x0 : 0.f);
                a[2 * j + 1] = (_Float16)((ka + 1 < Fin) ? x1 : 0.f);
            }
        }
        // B fragments: pre-packed per-lane; one 32B contiguous load each.
#pragma unroll
        for (int nt = 0; nt < NT; ++nt) {
            const int ct = ctBase + nt;
            const v16h b = *reinterpret_cast<const v16h*>(
                Bp + (((size_t)ks * nct + ct) * 32 + lane) * 16);
            c[nt] = __builtin_amdgcn_wmma_f32_16x16x32_f16(false, a, false, b,
                                                           (short)0, c[nt], false, false);
        }
    }

    // Write-only epilogue: C/D layout VGPR r -> M = r + 8*hi, N = lane&15.
#pragma unroll
    for (int nt = 0; nt < NT; ++nt) {
        const int col = (ctBase + nt) * 16 + ln;
        const float bv = bias[(!GN || col < Fout) ? col : 0];
        if (!GN || col < Fout) {
#pragma unroll
            for (int r = 0; r < 8; ++r) {
                int m = r + 8 * hi;
                float v = bv + c[nt][r];
                if (elu) v = v > 0.f ? v : (__expf(v) - 1.f);
                out[(size_t)(rowBase + m) * Fout + col] = v;
            }
        }
    }
}

// ---------------------------------------------------------------------------
// Host orchestration
// ---------------------------------------------------------------------------
extern "C" void kernel_launch(void* const* d_in, const int* in_sizes, int n_in,
                              void* d_out, int out_size, void* d_ws, size_t ws_size,
                              hipStream_t stream) {
    (void)in_sizes; (void)n_in; (void)out_size; (void)ws_size;

    static const int NS[5] = {16384, 4096, 1024, 256, 64};

    // ---- input pointers (setup_inputs dict order, lists flattened) ----
    const float* x = (const float*)d_in[0];
    const int*   edges[4];
    const int*   down_cols[4]; const float* down_vals[4];
    const int*   up_cols[4];   const float* up_vals[4];
    const float* enW[4]; const float* enB[4];
    for (int l = 0; l < 4; ++l) {
        edges[l]     = (const int*)d_in[1 + l];
        down_cols[l] = (const int*)d_in[9 + l];
        down_vals[l] = (const float*)d_in[13 + l];
        up_cols[l]   = (const int*)d_in[21 + l];
        up_vals[l]   = (const float*)d_in[25 + l];
        enW[l]       = (const float*)d_in[29 + l];
        enB[l]       = (const float*)d_in[33 + l];
    }
    const float* encW = (const float*)d_in[37];
    const float* encB = (const float*)d_in[38];
    const float* decW = (const float*)d_in[39];
    const float* decB = (const float*)d_in[40];
    const float* deW[5]; const float* deB[5];
    for (int i = 0; i < 5; ++i) {
        deW[i] = (const float*)d_in[41 + i];
        deB[i] = (const float*)d_in[46 + i];
    }

    // ---- workspace carve (4-byte units) ----
    float* w = (float*)d_ws;
    size_t off = 0;
    float* normb[4];  int* srcPb[4];  float* nrmPb[4];  int* rowptrb[4];
    for (int l = 0; l < 4; ++l) { normb[l]  = w + off;          off += (size_t)6 * NS[l]; }
    for (int l = 0; l < 4; ++l) { nrmPb[l]  = w + off;          off += (size_t)6 * NS[l]; }
    for (int l = 0; l < 4; ++l) { srcPb[l]  = (int*)(w + off);  off += (size_t)6 * NS[l]; }
    for (int l = 0; l < 4; ++l) { rowptrb[l] = (int*)(w + off); off += (size_t)NS[l] + 1; }
    float* deg = w + off; off += NS[0];          // reused as f32 degree buffer
    int* cnt   = (int*)(w + off); off += NS[0];  // reused as count / cursor buffer
    off = (off + 15) & ~(size_t)15;              // 64B-align the packed weights
    _Float16* packB = (_Float16*)(w + off); off += 300000;   // 600K halves (max 524288)
    float* txAll = w + off; off += (size_t)BB * 16384 * (KCH * 32);  // basis buffer
    const size_t CAP = (size_t)BB * 16384 * 32;  // largest [B,N,F] activation
    float* hA = w + off; off += CAP;
    float* hB = w + off; off += CAP;

    const int TPB = 256;

    // ---- per level: edge norms + CSR-by-destination (src/nrm in dst order) ----
    for (int l = 0; l < 4; ++l) {
        int N = NS[l], E = 6 * N;
        const int* src = edges[l]; const int* dst = src + E;
        unsigned gN = (unsigned)cdivl(N, TPB), gE = (unsigned)cdivl(E, TPB);
        k_zero_f<<<gN, TPB, 0, stream>>>(deg, N);
        k_deg<<<gE, TPB, 0, stream>>>(deg, src, E);
        k_edge_norm<<<gE, TPB, 0, stream>>>(normb[l], deg, src, dst, E);
        k_zero_i<<<gN, TPB, 0, stream>>>(cnt, N);
        k_cnt<<<gE, TPB, 0, stream>>>(cnt, dst, E);
        k_scan<<<1, 32, 0, stream>>>(cnt, rowptrb[l], N);
        k_copy_i<<<gN, TPB, 0, stream>>>(cnt, rowptrb[l], N);   // cursor = rowptr
        k_place<<<gE, TPB, 0, stream>>>(cnt, src, dst, normb[l], srcPb[l], nrmPb[l], E);
    }

    // ---- GEMM dispatch: pack weights, then compile-time tile width / guards ----
    auto gemm = [&](const float* A, const float* W, const float* bias, float* out,
                    long rows, int Fin, int Fout, int elu) {
        const int ksteps = (Fin + 31) / 32;
        const int nct = (Fout + 15) / 16;
        const long ptot = (long)ksteps * nct * 512;
        k_pack_w<<<(unsigned)cdivl(ptot, TPB), TPB, 0, stream>>>(packB, W, Fin, Fout,
                                                                 nct, ptot);
        const unsigned gx = (unsigned)(rows / 16);
        const bool gk = (Fin % 32) != 0;
        const bool gn = (Fout % 16) != 0;
        if (gk && gn) {
            dim3 gg(gx, (unsigned)nct);
            k_wmma_gemm<1, true, true><<<gg, 32, 0, stream>>>(A, packB, bias, out,
                                                              (int)rows, Fin, Fout, nct, elu);
        } else if (gk) {
            if (nct % 2 == 0) {
                dim3 gg(gx, (unsigned)(nct / 2));
                k_wmma_gemm<2, true, false><<<gg, 32, 0, stream>>>(A, packB, bias, out,
                                                                   (int)rows, Fin, Fout, nct, elu);
            } else {
                dim3 gg(gx, (unsigned)nct);
                k_wmma_gemm<1, true, false><<<gg, 32, 0, stream>>>(A, packB, bias, out,
                                                                   (int)rows, Fin, Fout, nct, elu);
            }
        } else if (gn) {
            dim3 gg(gx, (unsigned)nct);
            k_wmma_gemm<1, false, true><<<gg, 32, 0, stream>>>(A, packB, bias, out,
                                                               (int)rows, Fin, Fout, nct, elu);
        } else if (nct % 8 == 0) {
            dim3 gg(gx, (unsigned)(nct / 8));
            k_wmma_gemm<8, false, false><<<gg, 32, 0, stream>>>(A, packB, bias, out,
                                                                (int)rows, Fin, Fout, nct, elu);
        } else if (nct % 4 == 0) {
            dim3 gg(gx, (unsigned)(nct / 4));
            k_wmma_gemm<4, false, false><<<gg, 32, 0, stream>>>(A, packB, bias, out,
                                                                (int)rows, Fin, Fout, nct, elu);
        } else {
            dim3 gg(gx, (unsigned)(nct / 2));
            k_wmma_gemm<2, false, false><<<gg, 32, 0, stream>>>(A, packB, bias, out,
                                                                (int)rows, Fin, Fout, nct, elu);
        }
    };

    // ---- ChebConv: gather-only recurrence into concatenated basis, one GEMM
    //      with reference weights [K,Fin,Fout] viewed as [(K*Fin),Fout].
    //      Precondition: Tx0 (= conv input) already sits in block 0 of txAll.
    auto cheb = [&](float* out, int lvl, const float* W, const float* bias,
                    int Fin, int Fout, bool elu) {
        const int N = NS[lvl];
        const long rows = (long)BB * N;
        const int S = KCH * Fin;
        const unsigned gG = (unsigned)cdivl(rows * Fin, TPB);

        // Tx1 = L_hat * Tx0
        k_spmm_gather<<<gG, TPB, 0, stream>>>(txAll, rowptrb[lvl], srcPb[lvl], nrmPb[lvl],
                                              N, Fin, S, 0, Fin, -1, 1.f);
        // Tx_k = 2*L_hat*Tx_{k-1} - Tx_{k-2}   (subtraction fused)
        for (int k = 2; k < KCH; ++k) {
            k_spmm_gather<<<gG, TPB, 0, stream>>>(txAll, rowptrb[lvl], srcPb[lvl], nrmPb[lvl],
                                                  N, Fin, S, (k - 1) * Fin, k * Fin,
                                                  (k - 2) * Fin, 2.f);
        }
        // out = [Tx0|..|Tx5] @ Wstacked + bias (+ELU)
        gemm(txAll, W, bias, out, rows, S, Fout, elu ? 1 : 0);
    };

    auto pool = [&](const float* in, float* out, const int* cols, const float* vals,
                    int Nout, int Nin, int F, int So) {
        long tot = (long)BB * Nout * F;
        k_pool<<<(unsigned)cdivl(tot, TPB), TPB, 0, stream>>>(out, in, cols, vals,
                                                              Nout, Nin, F, So);
    };

    // ======================= Encoder =======================
    static const int enFin[4]  = {3, 32, 32, 32};
    static const int enFout[4] = {32, 32, 32, 64};
    // conv0 input: copy x into block 0 of basis buffer
    k_block_copy<<<(unsigned)cdivl((long)BB * NS[0] * 3, TPB), TPB, 0, stream>>>(
        txAll, x, (long)BB * NS[0], 3, KCH * 3, 0);
    for (int i = 0; i < 4; ++i) {
        cheb(hA, i, enW[i], enB[i], enFin[i], enFout[i], true);
        if (i < 3) {
            pool(hA, txAll, down_cols[i], down_vals[i], NS[i + 1], NS[i],
                 enFout[i], KCH * enFin[i + 1]);
        } else {
            pool(hA, hB, down_cols[i], down_vals[i], NS[i + 1], NS[i],
                 enFout[i], enFout[i]);    // contiguous [16, 4096] for dense
        }
    }
    // z = hB_flat @ enc_W + enc_b, written straight to d_out
    const long OUT_ELEMS = (long)BB * NS[0] * 3;               // 786432
    float* zOut = (float*)d_out + OUT_ELEMS;
    gemm(hB, encW, encB, zOut, BB, NS[4] * 64, 128, 0);

    // ======================= Decoder =======================
    gemm(zOut, decW, decB, hA, BB, 128, NS[4] * 64, 0);        // hA = [16, 64, 64]
    static const int deFin[4]  = {64, 64, 32, 32};
    static const int deFout[4] = {64, 32, 32, 32};
    for (int i = 0; i < 4; ++i) {
        int lvl = 3 - i;
        pool(hA, txAll, up_cols[lvl], up_vals[lvl], NS[lvl], NS[lvl + 1],
             deFin[i], KCH * deFin[i]);
        cheb(hA, lvl, deW[i], deB[i], deFin[i], deFout[i], true);
    }
    // final ChebConv 32 -> 3 at level 0, no ELU, written to d_out.
    k_block_copy<<<(unsigned)cdivl((long)BB * NS[0] * 32, TPB), TPB, 0, stream>>>(
        txAll, hA, (long)BB * NS[0], 32, KCH * 32, 0);
    cheb((float*)d_out, 0, deW[4], deB[4], 32, 3, false);
}
